// memristor_dense_39419209842709
// MI455X (gfx1250) — matrix operand order repaired and checked by reading, not compiled, not run on hardware
//
#include <hip/hip_runtime.h>
#include <hip/hip_bf16.h>

typedef __attribute__((ext_vector_type(16))) _Float16 v16h;
typedef __attribute__((ext_vector_type(8)))  _Float16 v8h;
typedef __attribute__((ext_vector_type(8)))  float    v8f;

#define P_ROWS  1024
#define N_IN    784
#define M_DIM   785          // n_in + bias row
#define I_PAD   800          // padded inner per term (25 * 32)
#define N_OUT   128
#define NK      8            // Taylor terms exp(t) ~ sum t^k/k!, k<8
#define KI      (NK * I_PAD) // 6400 total GEMM inner dim
#define SCALE_B 64.0f
#define V_REF_C 0.25f
#define G_MIN_C (1.0f / 983.3f)
#define G_MAX_C (1.0f / 281.3f)
#define N_AVG_C 2.132f

// ---------- kernel 0: seed max|weight| with the exact bias value 0.5 ----------
__global__ void k_init(unsigned* wsmax) {
    if (threadIdx.x == 0) *wsmax = __float_as_uint(0.5f);
}

// ---------- kernel 1: max|w| over w_pos, w_neg (bits-compare, nonneg) ----------
__global__ void k_maxred(const float* __restrict__ wp, const float* __restrict__ wn,
                         unsigned* wsmax, int n) {
    __shared__ float red[256];
    float m = 0.0f;
    for (int idx = blockIdx.x * blockDim.x + threadIdx.x; idx < 2 * n;
         idx += gridDim.x * blockDim.x) {
        float v = (idx < n) ? wp[idx] : wn[idx - n];
        m = fmaxf(m, fabsf(v));
    }
    red[threadIdx.x] = m;
    __syncthreads();
    for (int s = 128; s > 0; s >>= 1) {
        if (threadIdx.x < s) red[threadIdx.x] = fmaxf(red[threadIdx.x], red[threadIdx.x + s]);
        __syncthreads();
    }
    if (threadIdx.x == 0) atomicMax(wsmax, __float_as_uint(red[0]));
}

// ---------- kernel 2: Bt[jo][k*800+i] = (Gp*(4*mu_p)^k - Gn*(4*mu_n)^k)*64 ----------
__global__ void k_build_b(const float* __restrict__ wp, const float* __restrict__ wn,
                          const float* __restrict__ bp, const float* __restrict__ bn,
                          const float* __restrict__ npar, const unsigned* __restrict__ wsmax,
                          _Float16* __restrict__ Bt) {
    int gid = blockIdx.x * blockDim.x + threadIdx.x;
    if (gid >= I_PAD * N_OUT) return;
    int i  = gid % I_PAD;
    int jo = gid / I_PAD;
    _Float16* dst = Bt + (size_t)jo * KI + i;
    if (i >= M_DIM) {
        for (int k = 0; k < NK; ++k) dst[k * I_PAD] = (_Float16)0.0f;
        return;
    }
    float maxw = __uint_as_float(*wsmax);
    float gs   = (G_MAX_C - G_MIN_C) / maxw;
    float w_p  = (i < N_IN) ? wp[i * N_OUT + jo] : bp[jo];
    float w_n  = (i < N_IN) ? wn[i * N_OUT + jo] : bn[jo];
    float Gp = G_MIN_C + gs * w_p;
    float Gn = G_MIN_C + gs * w_n;
    float np_p = npar[i * (2 * N_OUT) + 2 * jo];
    float np_n = npar[i * (2 * N_OUT) + 2 * jo + 1];
    float mp4 = 4.0f * logf(np_p * (1.0f / N_AVG_C));
    float mn4 = 4.0f * logf(np_n * (1.0f / N_AVG_C));
    float pp = 1.0f, pn = 1.0f;
    for (int k = 0; k < NK; ++k) {
        dst[k * I_PAD] = (_Float16)((Gp * pp - Gn * pn) * SCALE_B);
        pp *= mp4;
        pn *= mn4;
    }
}

// ---------- kernel 3: A[p][k*800+i] = s * n_avg^e * (e/4)^k / k! ----------
__global__ void k_build_a(const float* __restrict__ x, _Float16* __restrict__ A) {
    int gid = blockIdx.x * blockDim.x + threadIdx.x;
    if (gid >= P_ROWS * I_PAD) return;
    int i = gid % I_PAD;
    int p = gid / I_PAD;
    _Float16* dst = A + (size_t)p * KI + i;
    float v;
    if (i < N_IN)       v = x[p * N_IN + i];
    else if (i == N_IN) v = 1.0f;   // bias word line
    else                v = 0.0f;   // padding
    float V = 0.5f * v;             // k_V * x = 2*V_ref*x
    if (!(V > 0.0f)) {
        for (int k = 0; k < NK; ++k) dst[k * I_PAD] = (_Float16)0.0f;
        return;
    }
    float e    = log2f(fmaxf(V, 1e-12f)) + 2.0f;     // log2(|V|/V_ref)
    float base = exp2f(e * log2f(N_AVG_C));          // n_avg^e
    float t    = e * 0.25f;
    const float inv[7] = {1.0f, 0.5f, 0.33333334f, 0.25f, 0.2f, 0.16666667f, 0.14285715f};
    float a = base;
    for (int k = 0; k < NK; ++k) {
        dst[k * I_PAD] = (_Float16)a;
        if (k < NK - 1) a *= t * inv[k];
    }
}

// ---------- kernel 4: WMMA GEMM, K split by Taylor term for 8x parallelism ----
// 1024 wave-tiles: (32 Mtiles x 4 Ntiles) x 8 K-segments of 800.
// Each wave: 32x32 f32 partial accumulated over 25 chunks of K=32.
__global__ __launch_bounds__(128) void k_gemm(const _Float16* __restrict__ A,
                                              const _Float16* __restrict__ Bt,
                                              float* __restrict__ Cpart) {
    int lane = threadIdx.x & 31;
    int wid  = threadIdx.x >> 5;
    int gidw = blockIdx.x * 4 + wid;       // 0..1023
    int tile = gidw & 127;                 // output tile
    int seg  = gidw >> 7;                  // Taylor-term segment 0..7
    int tn = (tile & 3) << 5;              // N base (32-wide)
    int tm = (tile >> 2) << 5;             // M base (32-tall)
    int kc0 = seg * I_PAD;                 // 800-wide K slab
    int kc1 = kc0 + I_PAD;

    v8f acc00 = {}, acc01 = {}, acc10 = {}, acc11 = {};

    // A fragment: lane<16 holds K{0..7,16..23}, lane>=16 holds K{8..15,24..31} of row lane&15
    const _Float16* a0 = A + (size_t)(tm + (lane & 15)) * KI + ((lane >> 4) << 3);
    const _Float16* a1 = a0 + (size_t)16 * KI;
    // B fragment: lane<16 holds K{0..7,16..23}, lane>=16 holds K{8..15,24..31} of column lane&15
    const _Float16* b0 = Bt + (size_t)(tn + (lane & 15)) * KI + ((lane >> 4) << 3);
    const _Float16* b1 = b0 + (size_t)16 * KI;

    union U { v16h v; v8h h[2]; };

    for (int kc = kc0; kc < kc1; kc += 32) {
        U af0, af1, bf0, bf1;
        af0.h[0] = *(const v8h*)(a0 + kc);      af0.h[1] = *(const v8h*)(a0 + kc + 16);
        af1.h[0] = *(const v8h*)(a1 + kc);      af1.h[1] = *(const v8h*)(a1 + kc + 16);
        bf0.h[0] = *(const v8h*)(b0 + kc);      bf0.h[1] = *(const v8h*)(b0 + kc + 16);
        bf1.h[0] = *(const v8h*)(b1 + kc);      bf1.h[1] = *(const v8h*)(b1 + kc + 16);

        acc00 = __builtin_amdgcn_wmma_f32_16x16x32_f16(false, af0.v, false, bf0.v,
                                                       (short)0, acc00, false, false);
        acc01 = __builtin_amdgcn_wmma_f32_16x16x32_f16(false, af0.v, false, bf1.v,
                                                       (short)0, acc01, false, false);
        acc10 = __builtin_amdgcn_wmma_f32_16x16x32_f16(false, af1.v, false, bf0.v,
                                                       (short)0, acc10, false, false);
        acc11 = __builtin_amdgcn_wmma_f32_16x16x32_f16(false, af1.v, false, bf1.v,
                                                       (short)0, acc11, false, false);
    }

    float* cseg = Cpart + (size_t)seg * (P_ROWS * N_OUT);
    int rowoff = (lane >> 4) << 3;   // D layout: lanes 16-31 hold M = v+8
    int col    = lane & 15;
    v8f accs[4] = {acc00, acc01, acc10, acc11};
#pragma unroll
    for (int f = 0; f < 4; ++f) {
        int fm = f >> 1, fn = f & 1;
        int r0 = tm + fm * 16 + rowoff;
        int c  = tn + fn * 16 + col;
#pragma unroll
        for (int v = 0; v < 8; ++v)
            cseg[(size_t)(r0 + v) * N_OUT + c] = accs[f][v];
    }
}

// ---------- kernel 5: deterministic reduce over the 8 K-segments + scaling ----
__global__ void k_reduce(const float* __restrict__ Cpart, const unsigned* __restrict__ wsmax,
                         float* __restrict__ out) {
    int idx = blockIdx.x * blockDim.x + threadIdx.x;
    if (idx >= P_ROWS * N_OUT) return;
    // y = C * V_ref * max_w / (SCALE_B * k_V * (G_MAX-G_MIN)), k_V = 2*V_ref
    float maxw  = __uint_as_float(*wsmax);
    float scale = 0.5f * maxw / (SCALE_B * (G_MAX_C - G_MIN_C));
    float s = 0.0f;
#pragma unroll
    for (int k = 0; k < NK; ++k) s += Cpart[(size_t)k * (P_ROWS * N_OUT) + idx];
    out[idx] = s * scale;
}

extern "C" void kernel_launch(void* const* d_in, const int* in_sizes, int n_in,
                              void* d_out, int out_size, void* d_ws, size_t ws_size,
                              hipStream_t stream) {
    const float* x    = (const float*)d_in[0];
    const float* wp   = (const float*)d_in[1];
    const float* wn   = (const float*)d_in[2];
    const float* bp   = (const float*)d_in[3];
    const float* bn   = (const float*)d_in[4];
    const float* npar = (const float*)d_in[5];
    float* out = (float*)d_out;

    unsigned* wsmax = (unsigned*)d_ws;
    char* base = (char*)d_ws + 16;
    _Float16* A  = (_Float16*)base;                                     // 13,107,200 B
    _Float16* Bt = (_Float16*)(base + (size_t)P_ROWS * KI * 2);         //  1,638,400 B
    float* Cpart = (float*)(base + (size_t)P_ROWS * KI * 2
                                 + (size_t)N_OUT * KI * 2);             //  4,194,304 B

    k_init<<<1, 1, 0, stream>>>(wsmax);
    k_maxred<<<208, 256, 0, stream>>>(wp, wn, wsmax, N_IN * N_OUT);
    k_build_b<<<(I_PAD * N_OUT + 255) / 256, 256, 0, stream>>>(wp, wn, bp, bn, npar, wsmax, Bt);
    k_build_a<<<(P_ROWS * I_PAD + 255) / 256, 256, 0, stream>>>(x, A);
    k_gemm<<<256, 128, 0, stream>>>(A, Bt, Cpart);
    k_reduce<<<(P_ROWS * N_OUT + 255) / 256, 256, 0, stream>>>(Cpart, wsmax, out);
}
